// HierarchicalSelfAttention_88330297410261
// MI455X (gfx1250) — matrix operand order repaired and checked by reading, not compiled, not run on hardware
//
#include <hip/hip_runtime.h>

#define DEV __device__ __forceinline__

typedef __attribute__((ext_vector_type(16))) __bf16 v16bf;
typedef __attribute__((ext_vector_type(8)))  float  v8f;

union Frag { v16bf f; unsigned u[8]; };

constexpr int  LLV = 3;
constexpr int  BB  = 2;
constexpr int  SS  = 2048;
constexpr int  DDIM = 512;
constexpr int  HH  = 8;
constexpr int  HDD = 64;
constexpr int  MM  = BB * SS;             // 4096 tokens
constexpr long MD  = (long)MM * DDIM;     // 2,097,152 elems
constexpr long D2  = (long)DDIM * DDIM;   // 262,144 elems

DEV unsigned short f2bf(float x) {
  unsigned u = __float_as_uint(x);
  u += 0x7fffu + ((u >> 16) & 1u);        // round-to-nearest-even
  return (unsigned short)(u >> 16);
}

DEV v8f vzero() { v8f z = {0.f,0.f,0.f,0.f,0.f,0.f,0.f,0.f}; return z; }

DEV v8f wmma_bf16(const Frag& a, const Frag& b, v8f c) {
  return __builtin_amdgcn_wmma_f32_16x16x32_bf16(
      /*neg_a=*/false, a.f, /*neg_b=*/false, b.f,
      /*c_mod=*/(short)0, c, /*reuse_a=*/false, /*reuse_b=*/false);
}

// A fragment: row-major A, tile base Ab (row 0 of 16-row tile), K window [kbase, kbase+32)
// ISA 16-bit A layout: lane m = lane&15, group g = lane>>4; K(v,t) = (v>>2)*16 + g*8 + (v&3)*2 + t
DEV void load_a(Frag& fr, const unsigned short* Ab, int lda, int kbase, int lane) {
  const int m = lane & 15, g = lane >> 4;
  const unsigned short* p = Ab + (long)m * lda + kbase + g * 8;
#pragma unroll
  for (int v = 0; v < 8; ++v) {
    const int k = ((v >> 2) << 4) + ((v & 3) << 1);
    fr.u[v] = *(const unsigned*)(p + k);
  }
}

// B fragment from row-major Bt [N, ldb] (B transposed): lane n = n0 + lane&15,
// K(v,t) = (v>>2)*16 + g*8 + (v&3)*2 + t  (same K order as A fragment)
DEV void load_b(Frag& fr, const unsigned short* Bt, long ldb, int n0, int kbase, int lane) {
  const int n = n0 + (lane & 15), g = lane >> 4;
  const unsigned short* p = Bt + (long)n * ldb + kbase + g * 8;
#pragma unroll
  for (int v = 0; v < 8; ++v) fr.u[v] = *(const unsigned*)(p + ((v >> 2) << 4) + ((v & 3) << 1));
}

// ---------------- conversion kernels ----------------
__global__ void k_f2bf(const float* __restrict__ in, unsigned short* __restrict__ out, int n) {
  for (int i = blockIdx.x * blockDim.x + threadIdx.x; i < n; i += gridDim.x * blockDim.x)
    out[i] = f2bf(in[i]);
}

// in: nb matrices [K,N] fp32 -> out: nb matrices [N,K] bf16 (transposed)
__global__ void k_convT(const float* __restrict__ in, unsigned short* __restrict__ out,
                        int K, int N, int n) {
  const int kn = K * N;
  for (int i = blockIdx.x * blockDim.x + threadIdx.x; i < n; i += gridDim.x * blockDim.x) {
    const int b = i / kn, r = i - b * kn;
    const int k = r / N, nn = r - k * N;
    out[(long)b * kn + (long)nn * K + k] = f2bf(in[i]);
  }
}

// ---------------- level projection + LayerNorm (wave = 16x64 tile, WG = 16x512 row block) ----
__global__ __launch_bounds__(256) void k_gemm_ln(
    const unsigned short* __restrict__ xbf, const unsigned short* __restrict__ WpT,
    const float* __restrict__ bp, const float* __restrict__ gamma,
    const float* __restrict__ beta, unsigned short* __restrict__ xn) {
  const int l = blockIdx.x >> 8;
  const int mblk = blockIdx.x & 255;
  const int t = threadIdx.x, wave = t >> 5, lane = t & 31;
  const int g = lane >> 4, ln = lane & 15;
  const unsigned short* Wl = WpT + (long)l * D2;
  const unsigned short* Ab = xbf + (long)mblk * 16 * DDIM;

  v8f acc[4];
#pragma unroll
  for (int tn = 0; tn < 4; ++tn) acc[tn] = vzero();

  for (int kb = 0; kb < DDIM; kb += 32) {
    Frag a; load_a(a, Ab, DDIM, kb, lane);
#pragma unroll
    for (int tn = 0; tn < 4; ++tn) {
      Frag b; load_b(b, Wl, DDIM, wave * 64 + tn * 16, kb, lane);
      acc[tn] = wmma_bf16(a, b, acc[tn]);
    }
  }
#pragma unroll
  for (int tn = 0; tn < 4; ++tn) {
    const int col = wave * 64 + tn * 16 + ln;
    const float bv = bp[l * DDIM + col];
#pragma unroll
    for (int v = 0; v < 8; ++v) acc[tn][v] += bv;
  }

  __shared__ float s_sum[16 * 8], s_sq[16 * 8], s_mu[16], s_rs[16];
  float ps[8], pq[8];
#pragma unroll
  for (int v = 0; v < 8; ++v) { ps[v] = 0.f; pq[v] = 0.f; }
#pragma unroll
  for (int tn = 0; tn < 4; ++tn)
#pragma unroll
    for (int v = 0; v < 8; ++v) { const float x = acc[tn][v]; ps[v] += x; pq[v] += x * x; }
#pragma unroll
  for (int v = 0; v < 8; ++v)
#pragma unroll
    for (int off = 1; off < 16; off <<= 1) {
      ps[v] += __shfl_xor(ps[v], off);
      pq[v] += __shfl_xor(pq[v], off);
    }
  if (ln == 0) {
#pragma unroll
    for (int v = 0; v < 8; ++v) {
      const int row = v + 8 * g;
      s_sum[row * 8 + wave] = ps[v];
      s_sq[row * 8 + wave]  = pq[v];
    }
  }
  __syncthreads();
  if (t < 16) {
    float su = 0.f, sq = 0.f;
#pragma unroll
    for (int w8 = 0; w8 < 8; ++w8) { su += s_sum[t * 8 + w8]; sq += s_sq[t * 8 + w8]; }
    const float mu = su * (1.f / DDIM);
    const float var = sq * (1.f / DDIM) - mu * mu;
    s_mu[t] = mu;
    s_rs[t] = rsqrtf(var + 1e-5f);
  }
  __syncthreads();
#pragma unroll
  for (int tn = 0; tn < 4; ++tn) {
    const int col = wave * 64 + tn * 16 + ln;
    const float ga = gamma[l * DDIM + col], be = beta[l * DDIM + col];
#pragma unroll
    for (int v = 0; v < 8; ++v) {
      const int row = v + 8 * g;
      const float val = (acc[tn][v] - s_mu[row]) * s_rs[row] * ga + be;
      xn[((long)l * MM + mblk * 16 + row) * DDIM + col] = f2bf(val);
    }
  }
}

// ---------------- generic GEMM + bias, 64x64 tile per wave ---------------------------------
// WG = 8 waves as 2(M) x 4(N) -> 128x256 tile; grid = (M/128, N/256) = (32, 2)
// modes: 0 = bf16 rowmajor, 1 = bf16 V-transposed [b,h,d,s], 2 = fp32 rowmajor
__global__ __launch_bounds__(256) void k_gemm_bias(
    const unsigned short* __restrict__ A, long aStride,
    const unsigned short* __restrict__ Bt, long ldb, int bColStride,
    const float* __restrict__ bias, void* __restrict__ outp, int mode, int nmat) {
  const int t = threadIdx.x, wave = t >> 5, lane = t & 31;
  const int g = lane >> 4, ln = lane & 15;
  const int wm = wave >> 2;                       // 0..1
  const int wn = wave & 3;                        // 0..3
  const int m0 = blockIdx.x * 128 + wm * 64;
  const int n0 = blockIdx.y * 256 + wn * 64;

  v8f acc[4][4];
#pragma unroll
  for (int mi = 0; mi < 4; ++mi)
#pragma unroll
    for (int ni = 0; ni < 4; ++ni) acc[mi][ni] = vzero();

  for (int im = 0; im < nmat; ++im) {
    const unsigned short* Ab = A + im * aStride + (long)m0 * DDIM;
    for (int kb = 0; kb < DDIM; kb += 32) {
      Frag af[4], bf[4];
#pragma unroll
      for (int mi = 0; mi < 4; ++mi) load_a(af[mi], Ab + (long)mi * 16 * DDIM, DDIM, kb, lane);
#pragma unroll
      for (int ni = 0; ni < 4; ++ni) load_b(bf[ni], Bt, ldb, n0 + ni * 16, im * bColStride + kb, lane);
#pragma unroll
      for (int mi = 0; mi < 4; ++mi)
#pragma unroll
        for (int ni = 0; ni < 4; ++ni) acc[mi][ni] = wmma_bf16(af[mi], bf[ni], acc[mi][ni]);
    }
  }
#pragma unroll
  for (int ni = 0; ni < 4; ++ni) {
    const int col = n0 + ni * 16 + ln;
    const float bv = bias[col];
#pragma unroll
    for (int mi = 0; mi < 4; ++mi) {
#pragma unroll
      for (int v = 0; v < 8; ++v) {
        const int row = m0 + mi * 16 + v + 8 * g;
        const float val = acc[mi][ni][v] + bv;
        if (mode == 0) {
          ((unsigned short*)outp)[(long)row * DDIM + col] = f2bf(val);
        } else if (mode == 2) {
          ((float*)outp)[(long)row * DDIM + col] = val;
        } else {                                  // V transposed per level: [b,h,d,s]
          const int bb = row >> 11, s = row & 2047, h = col >> 6, d = col & 63;
          ((unsigned short*)outp)[(((long)bb * HH + h) * HDD + d) * SS + s] = f2bf(val);
        }
      }
    }
  }
}

// ---------------- flash attention per (l,b,h), 16 queries per wave ----------------
__global__ __launch_bounds__(256) void k_attn(
    const unsigned short* __restrict__ Qb, const unsigned short* __restrict__ Kb,
    const unsigned short* __restrict__ Vt, const int* __restrict__ mask,
    unsigned short* __restrict__ att) {
  int idx = blockIdx.x;
  const int qblk = idx & 15; idx >>= 4;
  const int h = idx & 7; idx >>= 3;
  const int b = idx & 1; const int l = idx >> 1;
  const int t = threadIdx.x, wave = t >> 5, lane = t & 31;
  const int g = lane >> 4, ln = lane & 15;
  const int q0 = qblk * 128 + wave * 16;

  const long lb = (long)l * BB + b;
  const unsigned short* Qh = Qb + lb * SS * DDIM + h * HDD;
  const unsigned short* Kh = Kb + lb * SS * DDIM + h * HDD;
  const unsigned short* Vh = Vt + (lb * HH + h) * (long)HDD * SS;
  const int* mk = mask + b * SS;

  Frag qa0, qa1;
  load_a(qa0, Qh + (long)q0 * DDIM, DDIM, 0, lane);
  load_a(qa1, Qh + (long)q0 * DDIM, DDIM, 32, lane);

  __shared__ unsigned short pls[8][16 * 32];      // per-wave P staging (C-layout -> A-layout)
  unsigned short* myP = &pls[wave][0];

  float mrow[8], lrow[8];
  v8f acc[4];
#pragma unroll
  for (int tn = 0; tn < 4; ++tn) acc[tn] = vzero();
#pragma unroll
  for (int v = 0; v < 8; ++v) { mrow[v] = -3.0e38f; lrow[v] = 0.f; }

  for (int k0 = 0; k0 < SS; k0 += 32) {
    // prefetch next iteration's K rows and V columns (global_prefetch_b8)
    if (k0 + 32 < SS) {
      __builtin_prefetch(Kh + (long)(k0 + 32 + lane) * DDIM, 0, 0);
      __builtin_prefetch(Vh + (long)lane * SS + (k0 + 32), 0, 0);
      __builtin_prefetch(Vh + (long)(lane + 32) * SS + (k0 + 32), 0, 0);
    }
    v8f st[2];
#pragma unroll
    for (int half = 0; half < 2; ++half) {
      v8f s = vzero();
      Frag kb0, kb1;
      const unsigned short* Kt = Kh + (long)(k0 + half * 16) * DDIM;
      load_b(kb0, Kt, DDIM, 0, 0, lane);
      load_b(kb1, Kt, DDIM, 0, 32, lane);
      s = wmma_bf16(qa0, kb0, s);
      s = wmma_bf16(qa1, kb1, s);
      const int key = k0 + half * 16 + ln;
      const bool dead = (mk[key] == 0);
#pragma unroll
      for (int v = 0; v < 8; ++v) s[v] = dead ? -1.0e9f : s[v] * 0.125f;   // 1/sqrt(64)
      st[half] = s;
    }
    float rw[8];
#pragma unroll
    for (int v = 0; v < 8; ++v) {
      float a = fmaxf(st[0][v], st[1][v]);
#pragma unroll
      for (int off = 1; off < 16; off <<= 1) a = fmaxf(a, __shfl_xor(a, off));
      const float nm = fmaxf(mrow[v], a);
      rw[v] = __expf(mrow[v] - nm);
      mrow[v] = nm;
      lrow[v] *= rw[v];
    }
#pragma unroll
    for (int tn = 0; tn < 4; ++tn)
#pragma unroll
      for (int v = 0; v < 8; ++v) acc[tn][v] *= rw[v];
#pragma unroll
    for (int half = 0; half < 2; ++half)
#pragma unroll
      for (int v = 0; v < 8; ++v) st[half][v] = __expf(st[half][v] - mrow[v]);
#pragma unroll
    for (int v = 0; v < 8; ++v) {
      float su = st[0][v] + st[1][v];
#pragma unroll
      for (int off = 1; off < 16; off <<= 1) su += __shfl_xor(su, off);
      lrow[v] += su;
    }
    // stage P into LDS (row-major 16x32 bf16), reload in A-fragment layout (same-wave DS is in-order)
#pragma unroll
    for (int half = 0; half < 2; ++half)
#pragma unroll
      for (int v = 0; v < 8; ++v) {
        const int row = v + 8 * g;
        myP[row * 32 + half * 16 + ln] = f2bf(st[half][v]);
      }
    Frag pf;
    {
      const unsigned short* pp = myP + ln * 32 + g * 8;
#pragma unroll
      for (int v = 0; v < 8; ++v) {
        const int k = ((v >> 2) << 4) + ((v & 3) << 1);
        pf.u[v] = *(const unsigned*)(pp + k);
      }
    }
#pragma unroll
    for (int tn = 0; tn < 4; ++tn) {
      Frag vb;
      load_b(vb, Vh + (long)(tn * 16) * SS, SS, 0, k0, lane);
      acc[tn] = wmma_bf16(pf, vb, acc[tn]);
    }
  }
#pragma unroll
  for (int tn = 0; tn < 4; ++tn) {
    const int col = h * HDD + tn * 16 + ln;
#pragma unroll
    for (int v = 0; v < 8; ++v) {
      const int row = v + 8 * g;
      const float o = acc[tn][v] / lrow[v];
      att[(lb * SS + q0 + row) * DDIM + col] = f2bf(o);
    }
  }
}

// ---------------- launch ----------------
extern "C" void kernel_launch(void* const* d_in, const int* in_sizes, int n_in,
                              void* d_out, int out_size, void* d_ws, size_t ws_size,
                              hipStream_t stream) {
  (void)in_sizes; (void)n_in; (void)out_size; (void)ws_size;
  const float* x      = (const float*)d_in[0];
  const int*   mask   = (const int*)  d_in[1];
  const float* Wp     = (const float*)d_in[2];
  const float* bp     = (const float*)d_in[3];
  const float* gamma  = (const float*)d_in[4];
  const float* betaln = (const float*)d_in[5];
  const float* Wq     = (const float*)d_in[6];
  const float* bq     = (const float*)d_in[7];
  const float* Wk     = (const float*)d_in[8];
  const float* bk     = (const float*)d_in[9];
  const float* Wv     = (const float*)d_in[10];
  const float* bv     = (const float*)d_in[11];
  const float* Wo     = (const float*)d_in[12];
  const float* bo     = (const float*)d_in[13];
  const float* Wout   = (const float*)d_in[14];
  const float* bout   = (const float*)d_in[15];

  unsigned short* w = (unsigned short*)d_ws;
  unsigned short* xbf  = w; w += MD;
  unsigned short* WpT  = w; w += 3 * D2;
  unsigned short* WqT  = w; w += 3 * D2;
  unsigned short* WkT  = w; w += 3 * D2;
  unsigned short* WvT  = w; w += 3 * D2;
  unsigned short* WoT  = w; w += 3 * D2;
  unsigned short* WouT = w; w += 3 * D2;   // [512, 1536] transposed Wout
  unsigned short* xn   = w; w += 3 * MD;
  unsigned short* Qb   = w; w += 3 * MD;
  unsigned short* Kb   = w; w += 3 * MD;
  unsigned short* Vt   = w; w += 3 * MD;
  unsigned short* att  = xn;               // reuse: xn dead after QKV
  unsigned short* lev  = Qb;               // reuse: Q dead after attention

  const dim3 blk(256);
  const dim3 ggrid(32, 2);                 // 128x256 tile per WG
  k_f2bf<<<dim3((unsigned)(MD / 256)), blk, 0, stream>>>(x, xbf, (int)MD);
  k_convT<<<dim3(3072), blk, 0, stream>>>(Wp, WpT, DDIM, DDIM, (int)(3 * D2));
  k_convT<<<dim3(3072), blk, 0, stream>>>(Wq, WqT, DDIM, DDIM, (int)(3 * D2));
  k_convT<<<dim3(3072), blk, 0, stream>>>(Wk, WkT, DDIM, DDIM, (int)(3 * D2));
  k_convT<<<dim3(3072), blk, 0, stream>>>(Wv, WvT, DDIM, DDIM, (int)(3 * D2));
  k_convT<<<dim3(3072), blk, 0, stream>>>(Wo, WoT, DDIM, DDIM, (int)(3 * D2));
  k_convT<<<dim3(3072), blk, 0, stream>>>(Wout, WouT, 3 * DDIM, DDIM, (int)(3 * D2));

  k_gemm_ln<<<dim3(3 * 256), blk, 0, stream>>>(xbf, WpT, bp, gamma, betaln, xn);

  for (int l = 0; l < 3; ++l) {
    k_gemm_bias<<<ggrid, blk, 0, stream>>>(xn + l * MD, 0L, WqT + l * D2, (long)DDIM, 0,
                                           bq + l * DDIM, (void*)(Qb + l * MD), 0, 1);
    k_gemm_bias<<<ggrid, blk, 0, stream>>>(xn + l * MD, 0L, WkT + l * D2, (long)DDIM, 0,
                                           bk + l * DDIM, (void*)(Kb + l * MD), 0, 1);
    k_gemm_bias<<<ggrid, blk, 0, stream>>>(xn + l * MD, 0L, WvT + l * D2, (long)DDIM, 0,
                                           bv + l * DDIM, (void*)(Vt + l * MD), 1, 1);
  }

  k_attn<<<dim3(768), blk, 0, stream>>>(Qb, Kb, Vt, mask, att);

  for (int l = 0; l < 3; ++l)
    k_gemm_bias<<<ggrid, blk, 0, stream>>>(att + l * MD, 0L, WoT + l * D2, (long)DDIM, 0,
                                           bo + l * DDIM, (void*)(lev + l * MD), 0, 1);

  k_gemm_bias<<<ggrid, blk, 0, stream>>>(lev, MD, WouT, (long)(3 * DDIM), DDIM,
                                         bout, d_out, 2, 3);
}
